// MultiHeadLatentAttention_84473416778114
// MI455X (gfx1250) — matrix operand order repaired and checked by reading, not compiled, not run on hardware
//
#include <hip/hip_runtime.h>
#include <hip/hip_bf16.h>

// ---------------------------------------------------------------------------
// MLA (Multi-Head Latent Attention) for gfx1250 — all matmuls on
// v_wmma_f32_16x16x32_bf16 (bf16 in, fp32 accumulate), with async
// global->LDS tile staging (GLOBAL_LOAD_ASYNC_TO_LDS_B128) and
// double-buffered LDS to overlap copies with the WMMA stream.
// ---------------------------------------------------------------------------

typedef __attribute__((ext_vector_type(16))) __bf16 v16bf;
typedef __attribute__((ext_vector_type(8)))  __bf16 v8bf;
typedef __attribute__((ext_vector_type(8)))  float  v8f;

// Problem dims (fixed)
constexpr int BATCH = 2;
constexpr int SEQ   = 2048;
constexpr int DIN   = 2048;
constexpr int DOUT  = 2048;
constexpr int NH    = 16;
constexpr int DH    = 128;
constexpr int DCKV  = 512;
constexpr int DCQ   = 2048;
constexpr int MROWS = BATCH * SEQ;              // 4096
constexpr float INV_SCALE = 0.08838834764831845f; // 1/sqrt(128)

// GEMM tiling: block 128x128x32, 8 waves (2x4), wave tile 64x32 = 4x2 frags
constexpr int BM = 128, BN = 128, BK = 32;
constexpr int WM = 64,  WN = 32;
constexpr int FM = 4,   FN = 2;
constexpr int LDS_K = BK + 8;                   // 40 halves -> 80B row stride

__device__ __forceinline__ unsigned short f32_to_bf16(float f) {
  union { float f; unsigned int u; } v; v.f = f;
  unsigned int u = v.u;
  unsigned int r = (u + 0x7FFFu + ((u >> 16) & 1u)) >> 16;  // RNE
  return (unsigned short)r;
}

// Async copy of 16 bytes/lane from global to LDS (no VGPR staging).
// Low 32 bits of a generic LDS pointer are the LDS byte offset (aperture rule).
__device__ __forceinline__ void async_copy_b128(unsigned short* lds_dst,
                                                const unsigned short* gsrc) {
  unsigned lds = (unsigned)(uintptr_t)lds_dst;
  asm volatile("global_load_async_to_lds_b128 %0, %1, off"
               :: "v"(lds), "v"(gsrc) : "memory");
}
__device__ __forceinline__ void wait_async0() {
  asm volatile("s_wait_asynccnt 0x0" ::: "memory");
}

// ---------------------------------------------------------------------------
// Generic WMMA GEMM: C = A[M,K] * W[N,K]^T  (both row-major, K contiguous)
// Epilogues:
//  0: bf16 row-major            C16[m*ldc + n] = acc
//  1: q/k head scatter          dst[((b*NH+h)*SEQ + nr)*DH + dh]
//  2: v^T head scatter          dst[((b*NH+h)*DH + dh)*SEQ + nr]
//  3: scores: scale + causal    Cf[m*ldc + n] = (n<=m)? acc/sqrt(dh) : -inf
//  4: ctx head interleave       dst[((b*SEQ+m))*DOUT + h*DH + n]   (b,h from bh)
//  5: bias + fp32 out           Cf[m*ldc + n] = acc + bias[n]
// ---------------------------------------------------------------------------
template <int EPI>
__global__ void __launch_bounds__(256)
gemm_wmma(const unsigned short* __restrict__ A, int lda,
          const unsigned short* __restrict__ Bw, int ldb,
          void* __restrict__ Cptr, int ldc,
          int K, const float* __restrict__ bias, int bh)
{
  const int n0   = blockIdx.x * BN;
  const int m0   = blockIdx.y * BM;
  const int lane = threadIdx.x & 31;
  const int wave = threadIdx.x >> 5;
  const int wm   = (wave >> 2) * WM;   // 0 or 64
  const int wn   = (wave & 3) * WN;    // 0,32,64,96
  const int mj   = (lane >> 4) * 8;    // C-frag M offset from lane
  const int nl   = lane & 15;          // C-frag N offset from lane

  if constexpr (EPI == 3) {
    // Block fully above the causal diagonal: write -inf, skip all compute.
    if (n0 >= m0 + BM) {
      float* Cf = (float*)Cptr;
      #pragma unroll
      for (int fm = 0; fm < FM; ++fm)
        #pragma unroll
        for (int fn = 0; fn < FN; ++fn)
          #pragma unroll
          for (int j = 0; j < 8; ++j) {
            int m = m0 + wm + fm * 16 + mj + j;
            int n = n0 + wn + fn * 16 + nl;
            Cf[(size_t)m * ldc + n] = -__builtin_inff();
          }
      return;
    }
  }

  __shared__ unsigned short As[2][BM * LDS_K];
  __shared__ unsigned short Bs[2][BN * LDS_K];

  v8f acc[FM][FN];
  #pragma unroll
  for (int i = 0; i < FM; ++i)
    #pragma unroll
    for (int j = 0; j < FN; ++j)
      acc[i][j] = (v8f){};

  const int lr = threadIdx.x >> 1;        // row loaded by this thread (0..127)
  const int lc = (threadIdx.x & 1) * 16;  // half-col base (0 or 16)
  const int halfsel = lane >> 4;
  const int akb = halfsel * 8;            // A-frag K base (halves)
  const int bkb = halfsel * 8;            // B-frag K base (halves)

  const unsigned short* agRow = A  + (size_t)(m0 + lr) * lda + lc;
  const unsigned short* bgRow = Bw + (size_t)(n0 + lr) * ldb + lc;

  auto issue_tile = [&](int buf, int k0) {
    const unsigned short* ag = agRow + k0;
    async_copy_b128(&As[buf][lr * LDS_K + lc],     ag);
    async_copy_b128(&As[buf][lr * LDS_K + lc + 8], ag + 8);
    const unsigned short* bg = bgRow + k0;
    async_copy_b128(&Bs[buf][lr * LDS_K + lc],     bg);
    async_copy_b128(&Bs[buf][lr * LDS_K + lc + 8], bg + 8);
  };

  // Prologue: stage tile 0.
  issue_tile(0, 0);
  wait_async0();
  __syncthreads();

  int buf = 0;
  for (int k0 = 0; k0 < K; k0 += BK) {
    // Prefetch next tile into the other buffer while we compute this one.
    if (k0 + BK < K) issue_tile(buf ^ 1, k0 + BK);

    v16bf af[FM], bf[FN];
    #pragma unroll
    for (int i = 0; i < FM; ++i) {
      const unsigned short* p = &As[buf][(wm + i * 16 + nl) * LDS_K + akb];
      v8bf lo = *(const v8bf*)p;          // K = akb..akb+7
      v8bf hi = *(const v8bf*)(p + 16);   // K = akb+16..akb+23
      af[i] = __builtin_shufflevector(lo, hi, 0,1,2,3,4,5,6,7,8,9,10,11,12,13,14,15);
    }
    #pragma unroll
    for (int i = 0; i < FN; ++i) {
      const unsigned short* p = &Bs[buf][(wn + i * 16 + nl) * LDS_K + bkb];
      v8bf lo = *(const v8bf*)p;          // K = bkb..bkb+7
      v8bf hi = *(const v8bf*)(p + 16);   // K = bkb+16..bkb+23
      bf[i] = __builtin_shufflevector(lo, hi, 0,1,2,3,4,5,6,7,8,9,10,11,12,13,14,15);
    }
    #pragma unroll
    for (int i = 0; i < FM; ++i)
      #pragma unroll
      for (int j = 0; j < FN; ++j)
        acc[i][j] = __builtin_amdgcn_wmma_f32_16x16x32_bf16(
            false, af[i], false, bf[j], (short)0, acc[i][j], false, false);

    // Make this wave's prefetch visible, then sync the workgroup and swap.
    wait_async0();
    __syncthreads();
    buf ^= 1;
  }

  // Epilogue
  #pragma unroll
  for (int fm = 0; fm < FM; ++fm)
    #pragma unroll
    for (int fn = 0; fn < FN; ++fn)
      #pragma unroll
      for (int j = 0; j < 8; ++j) {
        const int m = m0 + wm + fm * 16 + mj + j;
        const int n = n0 + wn + fn * 16 + nl;
        const float v = acc[fm][fn][j];
        if constexpr (EPI == 0) {
          ((unsigned short*)Cptr)[(size_t)m * ldc + n] = f32_to_bf16(v);
        } else if constexpr (EPI == 1) {
          const int b = m >> 11, nr = m & (SEQ - 1);
          const int h = n >> 7,  dh = n & (DH - 1);
          ((unsigned short*)Cptr)[((size_t)(b * NH + h) * SEQ + nr) * DH + dh] = f32_to_bf16(v);
        } else if constexpr (EPI == 2) {
          const int b = m >> 11, nr = m & (SEQ - 1);
          const int h = n >> 7,  dh = n & (DH - 1);
          ((unsigned short*)Cptr)[((size_t)(b * NH + h) * DH + dh) * SEQ + nr] = f32_to_bf16(v);
        } else if constexpr (EPI == 3) {
          ((float*)Cptr)[(size_t)m * ldc + n] = (n <= m) ? v * INV_SCALE : -__builtin_inff();
        } else if constexpr (EPI == 4) {
          const int b = bh >> 4, h = bh & 15;
          ((unsigned short*)Cptr)[((size_t)(b * SEQ + m)) * DOUT + h * DH + n] = f32_to_bf16(v);
        } else {  // EPI == 5
          ((float*)Cptr)[(size_t)m * ldc + n] = v + bias[n];
        }
      }
}

// ---------------------------------------------------------------------------
// Row softmax over masked fp32 scores -> bf16 probabilities. One row/block.
// ---------------------------------------------------------------------------
__global__ void __launch_bounds__(256)
softmax_rows(const float* __restrict__ S, unsigned short* __restrict__ P)
{
  const int row = blockIdx.x;
  const float* s = S + (size_t)row * SEQ;
  unsigned short* p = P + (size_t)row * SEQ;
  __shared__ float red[256];
  const int t = threadIdx.x;

  float vals[8];
  float mx = -__builtin_inff();
  #pragma unroll
  for (int i = 0; i < 8; ++i) { vals[i] = s[t + i * 256]; mx = fmaxf(mx, vals[i]); }
  red[t] = mx; __syncthreads();
  #pragma unroll
  for (int off = 128; off > 0; off >>= 1) {
    if (t < off) red[t] = fmaxf(red[t], red[t + off]);
    __syncthreads();
  }
  mx = red[0]; __syncthreads();

  float sum = 0.f;
  #pragma unroll
  for (int i = 0; i < 8; ++i) { vals[i] = __expf(vals[i] - mx); sum += vals[i]; }
  red[t] = sum; __syncthreads();
  #pragma unroll
  for (int off = 128; off > 0; off >>= 1) {
    if (t < off) red[t] += red[t + off];
    __syncthreads();
  }
  const float inv = 1.0f / red[0];
  #pragma unroll
  for (int i = 0; i < 8; ++i) p[t + i * 256] = f32_to_bf16(vals[i] * inv);
}

// fp32 -> bf16, 4 elements/thread, counts are multiples of 1024
__global__ void __launch_bounds__(256)
cvt_f32_bf16(const float* __restrict__ in, unsigned short* __restrict__ out)
{
  const size_t i = ((size_t)blockIdx.x * 256 + threadIdx.x) * 4;
  float4 f = *(const float4*)(in + i);
  ushort4 o;
  o.x = f32_to_bf16(f.x); o.y = f32_to_bf16(f.y);
  o.z = f32_to_bf16(f.z); o.w = f32_to_bf16(f.w);
  *(ushort4*)(out + i) = o;
}

// ---------------------------------------------------------------------------
extern "C" void kernel_launch(void* const* d_in, const int* in_sizes, int n_in,
                              void* d_out, int out_size, void* d_ws, size_t ws_size,
                              hipStream_t stream)
{
  (void)in_sizes; (void)n_in; (void)out_size; (void)ws_size;
  const float* x     = (const float*)d_in[0];
  const float* W_DQ  = (const float*)d_in[1];
  const float* W_UQ  = (const float*)d_in[2];
  const float* W_DKV = (const float*)d_in[3];
  const float* W_UK  = (const float*)d_in[4];
  const float* W_UV  = (const float*)d_in[5];
  const float* W_O   = (const float*)d_in[6];
  const float* b_O   = (const float*)d_in[7];
  float* out = (float*)d_out;

  // Workspace carve-up (all 256B aligned)
  char* ws = (char*)d_ws;
  size_t off = 0;
  auto carve = [&](size_t bytes) {
    void* p = ws + off;
    off += (bytes + 255) & ~(size_t)255;
    return p;
  };
  unsigned short* xb   = (unsigned short*)carve((size_t)MROWS * DIN  * 2);
  unsigned short* wdq  = (unsigned short*)carve((size_t)DCQ   * DIN  * 2);
  unsigned short* wuq  = (unsigned short*)carve((size_t)DOUT  * DCQ  * 2);
  unsigned short* wdkv = (unsigned short*)carve((size_t)DCKV  * DIN  * 2);
  unsigned short* wuk  = (unsigned short*)carve((size_t)DOUT  * DCKV * 2);
  unsigned short* wuv  = (unsigned short*)carve((size_t)DOUT  * DCKV * 2);
  unsigned short* wo   = (unsigned short*)carve((size_t)DIN   * DOUT * 2);
  unsigned short* cq   = (unsigned short*)carve((size_t)MROWS * DCQ  * 2);
  unsigned short* ckv  = (unsigned short*)carve((size_t)MROWS * DCKV * 2);
  unsigned short* qh   = (unsigned short*)carve((size_t)MROWS * DOUT * 2);
  unsigned short* kh   = (unsigned short*)carve((size_t)MROWS * DOUT * 2);
  unsigned short* vt   = (unsigned short*)carve((size_t)MROWS * DOUT * 2);
  unsigned short* ctx  = (unsigned short*)carve((size_t)MROWS * DOUT * 2);
  float*          sc   = (float*)         carve((size_t)SEQ * SEQ * 4);
  unsigned short* attn = (unsigned short*)carve((size_t)SEQ * SEQ * 2);

  const dim3 blk(256);
  auto cvt = [&](const float* src, unsigned short* dst, size_t n) {
    cvt_f32_bf16<<<dim3((unsigned)(n / 1024)), blk, 0, stream>>>(src, dst);
  };
  cvt(x,     xb,   (size_t)MROWS * DIN);
  cvt(W_DQ,  wdq,  (size_t)DCQ * DIN);
  cvt(W_UQ,  wuq,  (size_t)DOUT * DCQ);
  cvt(W_DKV, wdkv, (size_t)DCKV * DIN);
  cvt(W_UK,  wuk,  (size_t)DOUT * DCKV);
  cvt(W_UV,  wuv,  (size_t)DOUT * DCKV);
  cvt(W_O,   wo,   (size_t)DIN * DOUT);

  // c_q = x W_DQ^T             [4096, 2048]
  gemm_wmma<0><<<dim3(DCQ / BN, MROWS / BM), blk, 0, stream>>>(
      xb, DIN, wdq, DIN, cq, DCQ, DIN, nullptr, 0);
  // q = c_q W_UQ^T -> head layout qh[b,h,n,dh]
  gemm_wmma<1><<<dim3(DOUT / BN, MROWS / BM), blk, 0, stream>>>(
      cq, DCQ, wuq, DCQ, qh, 0, DCQ, nullptr, 0);
  // c_kv = x W_DKV^T           [4096, 512]
  gemm_wmma<0><<<dim3(DCKV / BN, MROWS / BM), blk, 0, stream>>>(
      xb, DIN, wdkv, DIN, ckv, DCKV, DIN, nullptr, 0);
  // k = c_kv W_UK^T -> head layout kh[b,h,n,dh]
  gemm_wmma<1><<<dim3(DOUT / BN, MROWS / BM), blk, 0, stream>>>(
      ckv, DCKV, wuk, DCKV, kh, 0, DCKV, nullptr, 0);
  // v = c_kv W_UV^T -> transposed head layout vt[b,h,dh,n]
  gemm_wmma<2><<<dim3(DOUT / BN, MROWS / BM), blk, 0, stream>>>(
      ckv, DCKV, wuv, DCKV, vt, 0, DCKV, nullptr, 0);

  // Attention per (b,h), reusing score/prob buffers (in-order stream)
  for (int bh = 0; bh < BATCH * NH; ++bh) {
    const unsigned short* qp = qh + (size_t)bh * SEQ * DH;
    const unsigned short* kp = kh + (size_t)bh * SEQ * DH;
    const unsigned short* vp = vt + (size_t)bh * DH * SEQ;
    // scores = (q k^T)/sqrt(dh) with causal mask, fp32
    gemm_wmma<3><<<dim3(SEQ / BN, SEQ / BM), blk, 0, stream>>>(
        qp, DH, kp, DH, sc, SEQ, DH, nullptr, bh);
    softmax_rows<<<dim3(SEQ), blk, 0, stream>>>(sc, attn);
    // ctx = attn v  (via v^T), scattered into [b,n, h*128+dh]
    gemm_wmma<4><<<dim3(DH / BN, SEQ / BM), blk, 0, stream>>>(
        attn, SEQ, vp, SEQ, ctx, 0, SEQ, nullptr, bh);
  }

  // out = ctx W_O^T + b_O, fp32
  gemm_wmma<5><<<dim3(DIN / BN, MROWS / BM), blk, 0, stream>>>(
      ctx, DOUT, wo, DOUT, out, DIN, DOUT, b_O, 0);
}
